// MultiHeadAttention_58686433132652
// MI455X (gfx1250) — compile-verified, matrix operand order checked by reading, not run on hardware
//
#include <hip/hip_runtime.h>

#define B_ 4
#define S_ 2048
#define D_ 512
#define H_ 8
#define HD_ 64

typedef _Float16 v16h  __attribute__((ext_vector_type(16)));
typedef _Float16 half8 __attribute__((ext_vector_type(8)));
typedef float    v8f   __attribute__((ext_vector_type(8)));
typedef float    f4    __attribute__((ext_vector_type(4)));

static __device__ __forceinline__ v8f wmma_f16(v16h a, v16h b, v8f c) {
  // 8 args: (neg_a, A, neg_b, B, c_mod, C, reuse_a, reuse_b)
  return __builtin_amdgcn_wmma_f32_16x16x32_f16(false, a, false, b, (short)0, c,
                                                false, false);
}

// One WMMA fragment per lane = halves at [p..p+7] and [p+16..p+23] (two b128 loads).
static __device__ __forceinline__ v16h load_frag(const _Float16* p) {
  half8 lo = *(const half8*)(p);
  half8 hi = *(const half8*)(p + 16);
  return __builtin_shufflevector(lo, hi, 0, 1, 2, 3, 4, 5, 6, 7, 8, 9, 10, 11,
                                 12, 13, 14, 15);
}

// ---- LDS B-panel: 64 cols x 512 K halves = 64KB, 16B-chunk XOR swizzle ----
// chunk' = chunk ^ (row & 15)  ->  ds_load_b128 across lanes n=0..15 hits all
// 64 banks exactly once (conflict-free) with zero padding.
static __device__ __forceinline__ uint32_t swz_off(uint32_t row, uint32_t kbyte) {
  uint32_t c = (kbyte >> 4) ^ (row & 15u);
  return row * 1024u + (c << 4);
}

static __device__ __forceinline__ v16h lds_frag(const _Float16* sh, int row,
                                                int kbyte) {
  const char* base = (const char*)sh;
  half8 lo = *(const half8*)(base + swz_off((uint32_t)row, (uint32_t)kbyte));
  half8 hi = *(const half8*)(base + swz_off((uint32_t)row, (uint32_t)kbyte + 32u));
  return __builtin_shufflevector(lo, hi, 0, 1, 2, 3, 4, 5, 6, 7, 8, 9, 10, 11,
                                 12, 13, 14, 15);
}

// Cooperatively copy a contiguous 64KB panel (64 cols x 1024B) into LDS using
// gfx1250 async-to-LDS DMA; tracked by ASYNCcnt. Caller must __syncthreads().
static __device__ __forceinline__ void stage_panel(const _Float16* __restrict__ src,
                                                   _Float16* sh) {
  uint32_t base = (uint32_t)(uintptr_t)sh;  // flat shared addr[31:0] == LDS offset
  uint64_t g = (uint64_t)(uintptr_t)src;
  int tid = threadIdx.x;
#pragma unroll
  for (int r = 0; r < 16; ++r) {            // 256 thr * 16 * 16B = 64KB
    uint32_t c = (uint32_t)(r * 256 + tid); // 16B chunk id, coalesced
    uint32_t row = c >> 6;                  // 64 chunks per 1024B row
    uint32_t cw = c & 63u;
    uint32_t lds = base + row * 1024u + ((cw ^ (row & 15u)) << 4);
    uint64_t ga = g + ((uint64_t)c << 4);
    asm volatile("global_load_async_to_lds_b128 %0, %1, off"
                 :
                 : "v"(lds), "v"(ga)
                 : "memory");
  }
  asm volatile("s_wait_asynccnt 0x0" ::: "memory");
}

// ---------------- Weight transpose + fp32 -> f16 ----------------
__global__ __launch_bounds__(256) void transpose_w(const float* __restrict__ W,
                                                   _Float16* __restrict__ Wt) {
  int idx = blockIdx.x * 256 + threadIdx.x;   // contiguous read of W[k][n]
  int k = idx >> 9;
  int nn = idx & 511;
  Wt[(size_t)nn * D_ + k] = (_Float16)W[idx]; // Wt[n][k] = W[k][n]
}

// ---------------- QKV projection: Y = X * W + b (f16 out) ----------------
// transposed==0: dst[b][h][s][hd]   (Q, K)
// transposed==1: dst[b][h][hd][s]   (V stored transposed for P*V WMMA B-frags)
__global__ __launch_bounds__(256) void proj_kernel(
    const float* __restrict__ X, const _Float16* __restrict__ Wt,
    const float* __restrict__ bias, _Float16* __restrict__ dst, int transposed) {
  __shared__ __attribute__((aligned(16))) _Float16 bpanel[64 * 512];
  const int lane = threadIdx.x & 31;
  const int wave = threadIdx.x >> 5;
  const int n = lane & 15;
  const int g = lane >> 4;
  const int koff = g * 8;
  const int row0 = (blockIdx.x * 8 + wave) * 16;
  const int col0 = blockIdx.y * 64;

  stage_panel(Wt + (size_t)col0 * D_, bpanel);  // B panel shared by all 8 waves
  __syncthreads();

  v8f acc[4] = {};
  const float* arow = X + (size_t)(row0 + n) * D_;
#pragma unroll 4
  for (int k0 = 0; k0 < D_; k0 += 32) {
    const float* ap = arow + k0 + koff;
    f4 a0 = *(const f4*)(ap);
    f4 a1 = *(const f4*)(ap + 4);
    f4 a2 = *(const f4*)(ap + 16);
    f4 a3 = *(const f4*)(ap + 20);
    v16h af;
#pragma unroll
    for (int i = 0; i < 4; ++i) {
      af[i] = (_Float16)a0[i];
      af[4 + i] = (_Float16)a1[i];
      af[8 + i] = (_Float16)a2[i];
      af[12 + i] = (_Float16)a3[i];
    }
#pragma unroll
    for (int t = 0; t < 4; ++t)
      acc[t] = wmma_f16(af, lds_frag(bpanel, t * 16 + n, (k0 + koff) * 2),
                        acc[t]);
  }

#pragma unroll
  for (int t = 0; t < 4; ++t) {
    int col = col0 + t * 16 + n;
    int h = col >> 6, hd = col & 63;
    float bv = bias[col];
#pragma unroll
    for (int r = 0; r < 8; ++r) {
      int row = row0 + r + 8 * g;           // element (m = r + 8g, n = lane%16)
      int b = row >> 11, s = row & (S_ - 1);
      float val = acc[t][r] + bv;
      if (!transposed)
        dst[(((size_t)(b * H_ + h) * S_ + s) << 6) + hd] = (_Float16)val;
      else
        dst[((size_t)(b * H_ + h) * HD_ + hd) * S_ + s] = (_Float16)val;
    }
  }
}

// ---------------- Flash attention (online softmax, 1 wave = 16 queries) -----
__global__ __launch_bounds__(256) void attn_kernel(
    const _Float16* __restrict__ Qh, const _Float16* __restrict__ Kh,
    const _Float16* __restrict__ Vt, const int* __restrict__ valid_lens,
    _Float16* __restrict__ Om) {
  __shared__ __attribute__((aligned(16))) _Float16 plds[8 * 16 * 40]; // 40-half pitch
  const int lane = threadIdx.x & 31;
  const int wave = threadIdx.x >> 5;
  const int n = lane & 15;
  const int g = lane >> 4;
  const int koff = g * 8;
  const int bh = blockIdx.y;
  const int b = bh >> 3;
  const int h = bh & 7;
  const int vlen = valid_lens[b];
  const int q0 = blockIdx.x * 128 + wave * 16;

  const _Float16* Qrow = Qh + (((size_t)bh * S_ + q0 + n) << 6);
  v16h qa0 = load_frag(Qrow + koff);       // k = 0..31 pattern
  v16h qa1 = load_frag(Qrow + 32 + koff);  // k = 32..63 pattern

  v8f o[4] = {};
  float mrow[8], lrow[8];
#pragma unroll
  for (int r = 0; r < 8; ++r) {
    mrow[r] = -1e30f;
    lrow[r] = 0.f;
  }

  _Float16* pl = plds + wave * (16 * 40);

  for (int j0 = 0; j0 < S_; j0 += 32) {
    const _Float16* k1 = Kh + (((size_t)bh * S_ + j0 + n) << 6);
    const _Float16* k2 = k1 + (16 << 6);

    // scores = Q * K^T  (K rows are B-columns: lane n = key position)
    v8f s1 = {}, s2 = {};
    s1 = wmma_f16(qa0, load_frag(k1 + koff), s1);
    s1 = wmma_f16(qa1, load_frag(k1 + 32 + koff), s1);
    s2 = wmma_f16(qa0, load_frag(k2 + koff), s2);
    s2 = wmma_f16(qa1, load_frag(k2 + 32 + koff), s2);

    // Hoist V fragments: independent of softmax, loads overlap the VALU work.
    v16h vf[4];
#pragma unroll
    for (int t = 0; t < 4; ++t) {
      const _Float16* vr = Vt + ((size_t)bh * HD_ + t * 16 + n) * S_ + j0;
      vf[t] = load_frag(vr + koff);
      if (j0 + 32 < S_) __builtin_prefetch((const void*)(vr + 32), 0, 1);
    }
    if (j0 + 32 < S_) {                           // prefetch next key tile
      __builtin_prefetch((const void*)(k1 + (32 << 6)), 0, 1);
      __builtin_prefetch((const void*)(k2 + (32 << 6)), 0, 1);
    }

    const bool in1 = (j0 + n) < vlen;
    const bool in2 = (j0 + 16 + n) < vlen;

#pragma unroll
    for (int r = 0; r < 8; ++r) {
      float x1 = in1 ? s1[r] * 0.125f : -1.0e6f;  // 1/sqrt(64), then d2l mask
      float x2 = in2 ? s2[r] * 0.125f : -1.0e6f;
      float tmax = fmaxf(x1, x2);
#pragma unroll
      for (int off = 1; off < 16; off <<= 1)      // row-max across N lanes
        tmax = fmaxf(tmax, __shfl_xor(tmax, off, 32));
      float mnew = fmaxf(mrow[r], tmax);
      float alpha = __expf(mrow[r] - mnew);
      mrow[r] = mnew;
      float p1 = __expf(x1 - mnew);
      float p2 = __expf(x2 - mnew);
      float rs = p1 + p2;
#pragma unroll
      for (int off = 1; off < 16; off <<= 1)      // row-sum across N lanes
        rs += __shfl_xor(rs, off, 32);
      lrow[r] = lrow[r] * alpha + rs;
#pragma unroll
      for (int t = 0; t < 4; ++t) o[t][r] *= alpha;
      int m = r + 8 * g;                          // D-layout -> LDS row-major P
      pl[m * 40 + n] = (_Float16)p1;
      pl[m * 40 + 16 + n] = (_Float16)p2;
    }

    // P back as A-fragment (row m = lane%16, keys along K pattern)
    v16h pf = load_frag(pl + n * 40 + koff);
#pragma unroll
    for (int t = 0; t < 4; ++t)                   // O += P * V
      o[t] = wmma_f16(pf, vf[t], o[t]);
  }

#pragma unroll
  for (int r = 0; r < 8; ++r) {
    float inv = 1.0f / lrow[r];
    int s = q0 + r + 8 * g;
    _Float16* outp = Om + ((size_t)b * S_ + s) * D_ + h * HD_;
#pragma unroll
    for (int t = 0; t < 4; ++t)
      outp[t * 16 + n] = (_Float16)(o[t][r] * inv);
  }
}

// ---------------- Output projection: out = Om * Wo + bo (fp32 out) ----------
__global__ __launch_bounds__(256) void outproj_kernel(
    const _Float16* __restrict__ A, const _Float16* __restrict__ Wt,
    const float* __restrict__ bias, float* __restrict__ out) {
  __shared__ __attribute__((aligned(16))) _Float16 bpanel[64 * 512];
  const int lane = threadIdx.x & 31;
  const int wave = threadIdx.x >> 5;
  const int n = lane & 15;
  const int g = lane >> 4;
  const int koff = g * 8;
  const int row0 = (blockIdx.x * 8 + wave) * 16;
  const int col0 = blockIdx.y * 64;

  stage_panel(Wt + (size_t)col0 * D_, bpanel);
  __syncthreads();

  v8f acc[4] = {};
  const _Float16* arow = A + (size_t)(row0 + n) * D_;
#pragma unroll 4
  for (int k0 = 0; k0 < D_; k0 += 32) {
    v16h af = load_frag(arow + k0 + koff);
#pragma unroll
    for (int t = 0; t < 4; ++t)
      acc[t] = wmma_f16(af, lds_frag(bpanel, t * 16 + n, (k0 + koff) * 2),
                        acc[t]);
  }
#pragma unroll
  for (int t = 0; t < 4; ++t) {
    int col = col0 + t * 16 + n;
    float bv = bias[col];
#pragma unroll
    for (int r = 0; r < 8; ++r) {
      int row = row0 + r + 8 * g;
      out[(size_t)row * D_ + col] = acc[t][r] + bv;
    }
  }
}

extern "C" void kernel_launch(void* const* d_in, const int* in_sizes, int n_in,
                              void* d_out, int out_size, void* d_ws,
                              size_t ws_size, hipStream_t stream) {
  const float* q  = (const float*)d_in[0];
  const float* k  = (const float*)d_in[1];
  const float* v  = (const float*)d_in[2];
  const int*   vl = (const int*)d_in[3];
  const float* Wq = (const float*)d_in[4];
  const float* bq = (const float*)d_in[5];
  const float* Wk = (const float*)d_in[6];
  const float* bk = (const float*)d_in[7];
  const float* Wv = (const float*)d_in[8];
  const float* bv = (const float*)d_in[9];
  const float* Wo = (const float*)d_in[10];
  const float* bo = (const float*)d_in[11];
  float* out = (float*)d_out;

  _Float16* ws = (_Float16*)d_ws;
  const size_t WSZ = (size_t)D_ * D_;              // 256K halves per weight
  const size_t NH  = (size_t)B_ * H_ * S_ * HD_;   // 4M halves per tensor
  _Float16* WqT = ws;
  _Float16* WkT = WqT + WSZ;
  _Float16* WvT = WkT + WSZ;
  _Float16* WoT = WvT + WSZ;
  _Float16* Qh  = WoT + WSZ;
  _Float16* Kh  = Qh + NH;
  _Float16* Vt  = Kh + NH;
  _Float16* Om  = Vt + NH;

  transpose_w<<<1024, 256, 0, stream>>>(Wq, WqT);
  transpose_w<<<1024, 256, 0, stream>>>(Wk, WkT);
  transpose_w<<<1024, 256, 0, stream>>>(Wv, WvT);
  transpose_w<<<1024, 256, 0, stream>>>(Wo, WoT);

  dim3 gp(64, 8);
  proj_kernel<<<gp, 256, 0, stream>>>(q, WqT, bq, Qh, 0);
  proj_kernel<<<gp, 256, 0, stream>>>(k, WkT, bk, Kh, 0);
  proj_kernel<<<gp, 256, 0, stream>>>(v, WvT, bv, Vt, 1);

  attn_kernel<<<dim3(16, 32), 256, 0, stream>>>(Qh, Kh, Vt, vl, Om);

  outproj_kernel<<<gp, 256, 0, stream>>>(Om, WoT, bo, out);
}
